// OctonionTernaryLinearRef_28484223108016
// MI455X (gfx1250) — matrix operand order repaired and checked by reading, not compiled, not run on hardware
//
#include <hip/hip_runtime.h>
#include <hip/hip_bf16.h>

// ---------------------------------------------------------------------------
// OctonionTernaryLinear for MI455X (gfx1250, wave32, WMMA)
//   y[k,n,o] = beta[o] * sum_i sum_d  x[i,n,d] * SGN[k,i] * wq[IDX[k,i]][o,d]
//   wq = ternary(round(w/gamma) clipped to [-1,1]),  gamma = mean|w| + 1e-8
// Ternary weights are exact in bf16 -> v_wmma_f32_16x16x32_bf16, f32 accum.
// Compute-bound (~275 GFLOP vs ~280 MB): double-buffered LDS, async
// global->LDS B-tile copies (ASYNCcnt), register-staged A tile one step
// ahead, hardware bf16 converts, one barrier per K-step.
// ---------------------------------------------------------------------------

typedef __bf16 bf16_t;
typedef __attribute__((ext_vector_type(8)))  __bf16 v8bf;
typedef __attribute__((ext_vector_type(16))) __bf16 v16bf;
typedef __attribute__((ext_vector_type(8)))  float  v8f;

#define NTOK  8192
#define IN_O  512
#define OUT_O 512
#define NW    (8 * OUT_O * IN_O)   // 2,097,152 weight elements
#define KSTEPS 128                 // 8 i-chunks * (512/32)

__device__ __constant__ int c_IDX[64] = {
    0,1,2,3,4,5,6,7,
    1,0,3,2,5,4,7,6,
    2,3,0,1,6,7,4,5,
    3,2,1,0,7,6,5,4,
    4,5,6,7,0,1,2,3,
    5,4,7,6,1,0,3,2,
    6,7,4,5,2,3,0,1,
    7,6,5,4,3,2,1,0};
__device__ __constant__ float c_SGN[64] = {
     1,-1,-1,-1,-1,-1,-1,-1,
     1, 1, 1,-1, 1,-1,-1, 1,
     1,-1, 1, 1, 1, 1,-1,-1,
     1, 1,-1, 1, 1,-1, 1,-1,
     1,-1,-1,-1, 1, 1, 1, 1,
     1, 1,-1, 1,-1, 1,-1, 1,
     1, 1, 1,-1,-1, 1, 1,-1,
     1,-1, 1, 1,-1,-1, 1, 1};

// native RNE f32->bf16 convert (hardware v_cvt on gfx1250)
static __device__ inline void storeA16(unsigned short* ad, float s,
                                       float4 f0, float4 f1, float4 f2, float4 f3) {
    __bf16* bd = (__bf16*)ad;
    bd[0]  = (__bf16)(s * f0.x); bd[1]  = (__bf16)(s * f0.y);
    bd[2]  = (__bf16)(s * f0.z); bd[3]  = (__bf16)(s * f0.w);
    bd[4]  = (__bf16)(s * f1.x); bd[5]  = (__bf16)(s * f1.y);
    bd[6]  = (__bf16)(s * f1.z); bd[7]  = (__bf16)(s * f1.w);
    bd[8]  = (__bf16)(s * f2.x); bd[9]  = (__bf16)(s * f2.y);
    bd[10] = (__bf16)(s * f2.z); bd[11] = (__bf16)(s * f2.w);
    bd[12] = (__bf16)(s * f3.x); bd[13] = (__bf16)(s * f3.y);
    bd[14] = (__bf16)(s * f3.z); bd[15] = (__bf16)(s * f3.w);
}

// async copy of one 64-byte row: global -> LDS, tracked on ASYNCcnt.
// INST_OFFSET adds to BOTH the LDS and global address (ISA 15.18.3).
static __device__ inline void async_copy_row64(unsigned lds_dst,
                                               const unsigned short* gsrc) {
    unsigned long long ga = (unsigned long long)(size_t)gsrc;
    asm volatile(
        "global_load_async_to_lds_b128 %0, %1, off\n\t"
        "global_load_async_to_lds_b128 %0, %1, off offset:16\n\t"
        "global_load_async_to_lds_b128 %0, %1, off offset:32\n\t"
        "global_load_async_to_lds_b128 %0, %1, off offset:48"
        :: "v"(lds_dst), "v"(ga) : "memory");
}

static __device__ inline void wait_async0() {
    asm volatile("s_wait_asynccnt 0x0" ::: "memory");
}

// ---------------------------------------------------------------------------
// Phase 1: sum(|w|) reduction into ws[0]
// ---------------------------------------------------------------------------
__global__ __launch_bounds__(256) void absmean_sum(const float* __restrict__ w,
                                                   float* __restrict__ gsum, int n) {
    __shared__ float red[256];
    float s = 0.0f;
    for (int idx = blockIdx.x * 256 + threadIdx.x; idx < n; idx += gridDim.x * 256)
        s += fabsf(w[idx]);
    red[threadIdx.x] = s;
    __syncthreads();
    for (int off = 128; off > 0; off >>= 1) {
        if ((int)threadIdx.x < off) red[threadIdx.x] += red[threadIdx.x + off];
        __syncthreads();
    }
    if (threadIdx.x == 0) atomicAdd(gsum, red[0]);
}

// ---------------------------------------------------------------------------
// Phase 2: ternary quantize weights -> bf16 bits  (exact: values in {-1,0,1})
// ---------------------------------------------------------------------------
__global__ __launch_bounds__(256) void quantize_w(const float* __restrict__ w,
                                                  const float* __restrict__ gsum,
                                                  unsigned short* __restrict__ wqb, int n) {
    const float gamma = (*gsum) * (1.0f / (float)NW) + 1e-8f;
    const float inv = 1.0f / gamma;
    int idx = blockIdx.x * 256 + threadIdx.x;
    if (idx < n) {
        float q = rintf(w[idx] * inv);
        q = fminf(1.0f, fmaxf(-1.0f, q));
        __bf16 qb = (__bf16)q;                       // exact: {-1, 0, 1}
        wqb[idx] = __builtin_bit_cast(unsigned short, qb);
    }
}

// ---------------------------------------------------------------------------
// Phase 3: WMMA GEMM, 2-stage software pipeline.
//   Block: 256 threads (8 waves), output tile 128(M) x 256(N), full K=4096.
//   Wave (wm,wn) owns a 64x64 subtile = 4x4 WMMA 16x16 accumulators.
//   LDS: double-buffered A(8KB)+B(16KB) = 48 KB. One barrier per K-step.
//   grid = (512/256, 8192/128, 8) = (2, 64, 8)
// ---------------------------------------------------------------------------
__global__ __launch_bounds__(256, 1) void octo_gemm(
        const float* __restrict__ x,              // [8][8192][512] f32
        const unsigned short* __restrict__ wqb,   // [8][512][512] bf16 bits
        const float* __restrict__ beta,           // [512]
        float* __restrict__ y) {                  // [8][8192][512] f32

    __shared__ unsigned short lsA[2][128 * 32];   // [m][k] bf16
    __shared__ unsigned short lsB[2][256 * 32];   // [n][k] bf16

    const int ko = blockIdx.z;
    const int m0 = blockIdx.y * 128;
    const int n0 = blockIdx.x * 256;
    const int t  = threadIdx.x;
    const int wave = t >> 5;
    const int lane = t & 31;
    const int wm = wave >> 2;          // 0..1
    const int wn = wave & 3;           // 0..3
    const int lhalf = lane >> 4;       // 0 or 1
    const int lmod  = lane & 15;

    // A staging role: thread t loads 16 f32 of tile-row (t>>1), cols (t&1)*16..
    const int arow = t >> 1;
    const int acol = (t & 1) * 16;

    auto xptr = [&](int it) -> const float4* {
        const int i = it >> 4, d0 = (it & 15) * 32;
        return (const float4*)(x + ((size_t)i * NTOK + (size_t)(m0 + arow)) * IN_O
                               + acol + d0);
    };
    auto bptr = [&](int it) -> const unsigned short* {
        const int i = it >> 4, d0 = (it & 15) * 32;
        const int j = c_IDX[ko * 8 + i];
        return wqb + ((size_t)j * OUT_O + (size_t)(n0 + t)) * IN_O + d0;
    };

    v8f acc[4][4];
#pragma unroll
    for (int a = 0; a < 4; ++a)
#pragma unroll
        for (int b = 0; b < 4; ++b)
            acc[a][b] = v8f{0.f, 0.f, 0.f, 0.f, 0.f, 0.f, 0.f, 0.f};

    // ---- prologue: stage tile 0 into buffer 0 ----
    float4 fa0, fa1, fa2, fa3;
    {
        const float4* xp = xptr(0);
        fa0 = xp[0]; fa1 = xp[1]; fa2 = xp[2]; fa3 = xp[3];
        async_copy_row64((unsigned)(size_t)&lsB[0][t * 32], bptr(0));
        storeA16(&lsA[0][arow * 32 + acol], c_SGN[ko * 8], fa0, fa1, fa2, fa3);
        wait_async0();
        __syncthreads();
    }

    for (int it = 0; it < KSTEPS; ++it) {
        const int cur = it & 1;
        const int nxt = cur ^ 1;
        const bool more = (it + 1) < KSTEPS;

        // ---- issue next tile's loads (overlap with this step's WMMAs) ----
        if (more) {
            const float4* xp = xptr(it + 1);
            fa0 = xp[0]; fa1 = xp[1]; fa2 = xp[2]; fa3 = xp[3];
            async_copy_row64((unsigned)(size_t)&lsB[nxt][t * 32], bptr(it + 1));
            __builtin_prefetch(xptr(it + 2), 0, 1);   // speculative, 2 ahead
        }

        // ---- B fragments (contiguous k, same order as A): lane<16 -> col=lane K0..15;
        //                                lane>=16 -> col=lane-16 K16..31 ----
        v16bf bfr[4];
#pragma unroll
        for (int nt = 0; nt < 4; ++nt)
            bfr[nt] = *(const v16bf*)
                &lsB[cur][(wn * 64 + nt * 16 + lmod) * 32 + lhalf * 16];

        // stage next A tile into the other buffer (no reader this step)
        if (more)
            storeA16(&lsA[nxt][arow * 32 + acol],
                     c_SGN[ko * 8 + ((it + 1) >> 4)], fa0, fa1, fa2, fa3);

        // ---- 16 WMMAs: A frag per mt row (lane<16: row=lane, K0..15;
        //                lane>=16: row=lane-16, K16..31) ----
#pragma unroll
        for (int mt = 0; mt < 4; ++mt) {
            const unsigned short* pa =
                &lsA[cur][(wm * 64 + mt * 16 + lmod) * 32 + lhalf * 16];
            v8bf lo = *(const v8bf*)pa;
            v8bf hi = *(const v8bf*)(pa + 8);
            v16bf af = __builtin_shufflevector(lo, hi, 0, 1, 2, 3, 4, 5, 6, 7,
                                               8, 9, 10, 11, 12, 13, 14, 15);
#pragma unroll
            for (int nt = 0; nt < 4; ++nt)
                acc[mt][nt] = __builtin_amdgcn_wmma_f32_16x16x32_bf16(
                    false, af, false, bfr[nt],
                    (short)0, acc[mt][nt], false, false);
        }

        if (more) wait_async0();   // next buffer's B rows landed in LDS
        __syncthreads();           // everyone done reading cur / writing nxt
    }

    // ---- epilogue: C layout lane<16 -> N=lane, M=r; lane>=16 -> M=r+8 ----
#pragma unroll
    for (int mt = 0; mt < 4; ++mt) {
#pragma unroll
        for (int nt = 0; nt < 4; ++nt) {
            const int gm = m0 + wm * 64 + mt * 16;
            const int gn = n0 + wn * 64 + nt * 16 + lmod;
            const float bt = beta[gn];
            float* yp = y + ((size_t)ko * NTOK + (size_t)gm) * OUT_O + gn;
            const int rofs = lhalf * 8;
#pragma unroll
            for (int r = 0; r < 8; ++r)
                yp[(size_t)(r + rofs) * OUT_O] = acc[mt][nt][r] * bt;
        }
    }
}

// ---------------------------------------------------------------------------
extern "C" void kernel_launch(void* const* d_in, const int* in_sizes, int n_in,
                              void* d_out, int out_size, void* d_ws, size_t ws_size,
                              hipStream_t stream) {
    const float* x    = (const float*)d_in[0];   // [8][8192][512]
    const float* w    = (const float*)d_in[1];   // [8][512][512]
    const float* beta = (const float*)d_in[2];   // [512]
    float* y = (float*)d_out;                    // [8][8192][512]

    float* gsum = (float*)d_ws;
    unsigned short* wqb = (unsigned short*)((char*)d_ws + 256);  // 4 MB bf16 weights

    hipMemsetAsync(gsum, 0, sizeof(float), stream);
    absmean_sum<<<512, 256, 0, stream>>>(w, gsum, NW);
    quantize_w<<<(NW + 255) / 256, 256, 0, stream>>>(w, gsum, wqb, NW);

    dim3 grid(OUT_O / 256, NTOK / 128, 8);   // (2, 64, 8)
    octo_gemm<<<grid, 256, 0, stream>>>(x, wqb, beta, y);
}
